// MultiHeadAttention_1460288880893
// MI455X (gfx1250) — compile-verified, matrix operand order checked by reading, not run on hardware
//
#include <hip/hip_runtime.h>

typedef __attribute__((ext_vector_type(16))) _Float16 v16h;
typedef __attribute__((ext_vector_type(8)))  _Float16 v8h;
typedef __attribute__((ext_vector_type(8)))  float    v8f;
typedef __attribute__((ext_vector_type(4)))  float    v4f;

#define NUM_HEADS 16
#define SEQ       2048
#define DMODEL    1024
#define DHEAD     64
#define BATCHN    2

// ---------------------------------------------------------------------------
// WMMA wrapper: D = A(16x32 f16) * B(32x16 f16) + C(16x16 f32)
// ---------------------------------------------------------------------------
__device__ __forceinline__ v8f wmma_f16(v16h a, v16h b, v8f c) {
  return __builtin_amdgcn_wmma_f32_16x16x32_f16(
      /*neg_a=*/false, a, /*neg_b=*/false, b,
      /*c_mod=*/(short)0, c, /*reuse_a=*/false, /*reuse_b=*/false);
}

// ---------------------------------------------------------------------------
// Fragment loader. CDNA5 16-bit A-matrix layout (16x32, MxK):
//   lane = h*16 + r  (r = row M, h = K-half selector)
//   element j (0..15):  K = (j < 8) ? (j + 8*h) : (j + 8 + 8*h)
// => two contiguous 8-half (16B) runs at k-offsets (8h) and (16+8h).
// B fragments (32x16, KxN): lane r holds column N=r with the same K packing,
// so loading row N of the transposed operand with the same runs gives B.
// ---------------------------------------------------------------------------
__device__ __forceinline__ v16h frag_from_f16(const _Float16* __restrict__ p, int h) {
  v8h lo = *(const v8h*)(p + 8 * h);
  v8h hi = *(const v8h*)(p + 16 + 8 * h);
  v16h f;
#pragma unroll
  for (int i = 0; i < 8; ++i) {
    f[i]     = lo[i];
    f[8 + i] = hi[i];
  }
  return f;
}

// ---------------------------------------------------------------------------
// Kernel 0: one-shot f32 -> f16 conversion (8 elements / thread, b128 I/O).
// Amortizes the precision change so GEMM inner loops carry zero cvt VALU.
// ---------------------------------------------------------------------------
__global__ __launch_bounds__(256) void cvt_f16_kernel(
    const float* __restrict__ src, _Float16* __restrict__ dst, int n8) {
  const int i = blockIdx.x * blockDim.x + threadIdx.x;
  if (i >= n8) return;
  const v4f* p = (const v4f*)src + 2 * (size_t)i;
  v4f a = p[0], b = p[1];
  v8h o;
#pragma unroll
  for (int j = 0; j < 4; ++j) {
    o[j]     = (_Float16)a[j];
    o[4 + j] = (_Float16)b[j];
  }
  ((v8h*)dst)[i] = o;
}

// ---------------------------------------------------------------------------
// Kernel 1: projection GEMM  Out = X @ W^T   (f16 x f16 -> f16, scattered)
//   Xh: [BATCHN*SEQ, DMODEL] f16   Wh: [DMODEL, DMODEL] f16 (row = out feat)
// Output scatter:
//   v_transposed==0 : Out[((b*H+h)*SEQ + s)*DHEAD + d]      (Q, K layout)
//   v_transposed==1 : Out[((b*H+h)*DHEAD + d)*SEQ + s]      (V^T layout)
// Block = 256 threads = 8 waves; block tile 128x128; wave tile 32x64.
// ---------------------------------------------------------------------------
__global__ __launch_bounds__(256) void proj_gemm_kernel(
    const _Float16* __restrict__ X, const _Float16* __restrict__ W,
    _Float16* __restrict__ Out, int v_transposed) {
  const int lane = threadIdx.x & 31;
  const int wave = threadIdx.x >> 5;
  const int r    = lane & 15;
  const int hh   = lane >> 4;
  const int Mb   = blockIdx.y * 128 + (wave >> 1) * 32;
  const int Nb   = blockIdx.x * 128 + (wave & 1) * 64;

  const _Float16* xr0 = X + (size_t)(Mb + r) * DMODEL;
  const _Float16* xr1 = X + (size_t)(Mb + 16 + r) * DMODEL;
  const _Float16* wr[4];
#pragma unroll
  for (int j = 0; j < 4; ++j) wr[j] = W + (size_t)(Nb + 16 * j + r) * DMODEL;

  v8f zero = {};
  v8f acc[2][4];
#pragma unroll
  for (int i = 0; i < 2; ++i)
#pragma unroll
    for (int j = 0; j < 4; ++j) acc[i][j] = zero;

  for (int kb = 0; kb < DMODEL; kb += 32) {
    if (kb + 32 < DMODEL) {
      __builtin_prefetch(xr0 + kb + 32, 0, 1);
      __builtin_prefetch(wr[0] + kb + 32, 0, 1);
    }
    v16h a[2], b[4];
    a[0] = frag_from_f16(xr0 + kb, hh);
    a[1] = frag_from_f16(xr1 + kb, hh);
#pragma unroll
    for (int j = 0; j < 4; ++j) b[j] = frag_from_f16(wr[j] + kb, hh);
#pragma unroll
    for (int i = 0; i < 2; ++i)
#pragma unroll
      for (int j = 0; j < 4; ++j)
        acc[i][j] = wmma_f16(a[i], b[j], acc[i][j]);
  }

  // C/D layout: element g of the v8f lives at row (g + 8*hh), col r.
#pragma unroll
  for (int i = 0; i < 2; ++i)
#pragma unroll
    for (int j = 0; j < 4; ++j)
#pragma unroll
      for (int g = 0; g < 8; ++g) {
        const int m    = Mb + 16 * i + g + 8 * hh;
        const int n    = Nb + 16 * j + r;
        const int bidx = m >> 11;            // m / SEQ
        const int s    = m & (SEQ - 1);
        const int hidx = n >> 6;             // n / DHEAD
        const int d    = n & (DHEAD - 1);
        const size_t bh = (size_t)bidx * NUM_HEADS + hidx;
        const size_t dst = v_transposed
                               ? (bh * DHEAD + d) * SEQ + s
                               : (bh * SEQ + s) * DHEAD + d;
        Out[dst] = (_Float16)acc[i][j][g];
      }
}

// ---------------------------------------------------------------------------
// Kernel 2: causal flash attention, one wave per 16-row Q tile.
//   Qw,Kw: [B*H, SEQ, DHEAD] f16      Vt: [B*H, DHEAD, SEQ] f16
//   Attn : [B, SEQ, DMODEL] f16 (head-concatenated)
// Online softmax in f32; mask (-1e9) applied BEFORE 1/sqrt(d) scale, as in
// the reference. P tile is re-laid-out via per-wave LDS (same-wave DS ops are
// in-order on CDNA5, so only a compiler barrier is needed).
// ---------------------------------------------------------------------------
__global__ __launch_bounds__(128) void flash_attn_kernel(
    const _Float16* __restrict__ Qw, const _Float16* __restrict__ Kw,
    const _Float16* __restrict__ Vt, _Float16* __restrict__ Attn) {
  __shared__ _Float16 lds_p[4][16 * 32];

  const int lane  = threadIdx.x & 31;
  const int wave  = threadIdx.x >> 5;
  const int r     = lane & 15;
  const int hh    = lane >> 4;
  const int bh    = blockIdx.y;                 // b*H + h
  const int bidx  = bh >> 4;
  const int hidx  = bh & (NUM_HEADS - 1);
  const int qbase = (blockIdx.x * 4 + wave) * 16;

  const _Float16* Qp = Qw + (size_t)bh * SEQ * DHEAD;
  const _Float16* Kp = Kw + (size_t)bh * SEQ * DHEAD;
  const _Float16* Vp = Vt + (size_t)bh * DHEAD * SEQ;
  _Float16* pl = lds_p[wave];

  // Q fragments held in registers for the whole key loop (d_head = 64 = 2x32)
  v16h qf[2];
#pragma unroll
  for (int ks = 0; ks < 2; ++ks)
    qf[ks] = frag_from_f16(Qp + (size_t)(qbase + r) * DHEAD + 32 * ks, hh);

  v8f zero = {};
  v8f oacc[4];
#pragma unroll
  for (int nt = 0; nt < 4; ++nt) oacc[nt] = zero;

  float m_run[8], l_run[8];
#pragma unroll
  for (int g = 0; g < 8; ++g) { m_run[g] = -3.0e30f; l_run[g] = 0.0f; }

  const int nsteps = (qbase + 16 + 31) / 32;   // causal: keys <= qbase+15
  for (int t32 = 0; t32 < nsteps; ++t32) {
    const int kb = t32 * 32;

    // ---- scores: S[16x32] = Q[16x64] * K^T, two 16x16 WMMA column tiles ----
    v8f sc[2];
#pragma unroll
    for (int t = 0; t < 2; ++t) {
      sc[t] = zero;
#pragma unroll
      for (int ks = 0; ks < 2; ++ks) {
        v16h bk = frag_from_f16(Kp + (size_t)(kb + 16 * t + r) * DHEAD + 32 * ks, hh);
        sc[t] = wmma_f16(qf[ks], bk, sc[t]);
      }
    }

    // ---- causal mask (before scale) + 1/sqrt(64) scale ----
#pragma unroll
    for (int t = 0; t < 2; ++t)
#pragma unroll
      for (int g = 0; g < 8; ++g) {
        const int q   = qbase + g + 8 * hh;
        const int key = kb + 16 * t + r;
        float s = sc[t][g];
        s = (key > q) ? -1.0e9f : s;
        sc[t][g] = s * 0.125f;
      }

    // ---- online softmax: row max over 32 cols (16 lanes per half-wave) ----
    float mx[8];
#pragma unroll
    for (int g = 0; g < 8; ++g) mx[g] = fmaxf(sc[0][g], sc[1][g]);
#pragma unroll
    for (int off = 1; off < 16; off <<= 1)
#pragma unroll
      for (int g = 0; g < 8; ++g) mx[g] = fmaxf(mx[g], __shfl_xor(mx[g], off, 32));

    float alpha[8];
#pragma unroll
    for (int g = 0; g < 8; ++g) {
      const float mn = fmaxf(m_run[g], mx[g]);
      alpha[g] = __expf(m_run[g] - mn);
      m_run[g] = mn;
    }
#pragma unroll
    for (int t = 0; t < 2; ++t)
#pragma unroll
      for (int g = 0; g < 8; ++g) sc[t][g] = __expf(sc[t][g] - m_run[g]);

    float rs[8];
#pragma unroll
    for (int g = 0; g < 8; ++g) rs[g] = sc[0][g] + sc[1][g];
#pragma unroll
    for (int off = 1; off < 16; off <<= 1)
#pragma unroll
      for (int g = 0; g < 8; ++g) rs[g] += __shfl_xor(rs[g], off, 32);
#pragma unroll
    for (int g = 0; g < 8; ++g) l_run[g] = l_run[g] * alpha[g] + rs[g];
#pragma unroll
    for (int nt = 0; nt < 4; ++nt)
#pragma unroll
      for (int g = 0; g < 8; ++g) oacc[nt][g] *= alpha[g];

    // ---- bounce P (C/D layout) through LDS into A-fragment layout ----
#pragma unroll
    for (int t = 0; t < 2; ++t)
#pragma unroll
      for (int g = 0; g < 8; ++g)
        pl[(g + 8 * hh) * 32 + 16 * t + r] = (_Float16)sc[t][g];
    asm volatile("" ::: "memory");                 // order DS write -> DS read
    v16h pf = frag_from_f16(pl + r * 32, hh);      // rows = r, 32 key cols
    asm volatile("" ::: "memory");

    // ---- O[16x64] += P[16x32] * V[32x64] : 4 WMMAs ----
#pragma unroll
    for (int nt = 0; nt < 4; ++nt) {
      v16h bv = frag_from_f16(Vp + (size_t)(16 * nt + r) * SEQ + kb, hh);
      oacc[nt] = wmma_f16(pf, bv, oacc[nt]);
    }
  }

  // ---- normalize and store head-concatenated f16 attention output ----
  float inv[8];
#pragma unroll
  for (int g = 0; g < 8; ++g) inv[g] = 1.0f / l_run[g];
#pragma unroll
  for (int nt = 0; nt < 4; ++nt)
#pragma unroll
    for (int g = 0; g < 8; ++g) {
      const int q = qbase + g + 8 * hh;
      const int d = 16 * nt + r;
      Attn[((size_t)bidx * SEQ + q) * DMODEL + hidx * DHEAD + d] =
          (_Float16)(oacc[nt][g] * inv[g]);
    }
}

// ---------------------------------------------------------------------------
// Kernel 3: output projection  Out = Attn @ Wo^T  (f16 x f16 -> f32)
// ---------------------------------------------------------------------------
__global__ __launch_bounds__(256) void out_gemm_kernel(
    const _Float16* __restrict__ A, const _Float16* __restrict__ W,
    float* __restrict__ Out) {
  const int lane = threadIdx.x & 31;
  const int wave = threadIdx.x >> 5;
  const int r    = lane & 15;
  const int hh   = lane >> 4;
  const int Mb   = blockIdx.y * 128 + (wave >> 1) * 32;
  const int Nb   = blockIdx.x * 128 + (wave & 1) * 64;

  const _Float16* ar0 = A + (size_t)(Mb + r) * DMODEL;
  const _Float16* ar1 = A + (size_t)(Mb + 16 + r) * DMODEL;
  const _Float16* wr[4];
#pragma unroll
  for (int j = 0; j < 4; ++j) wr[j] = W + (size_t)(Nb + 16 * j + r) * DMODEL;

  v8f zero = {};
  v8f acc[2][4];
#pragma unroll
  for (int i = 0; i < 2; ++i)
#pragma unroll
    for (int j = 0; j < 4; ++j) acc[i][j] = zero;

  for (int kb = 0; kb < DMODEL; kb += 32) {
    if (kb + 32 < DMODEL) {
      __builtin_prefetch(ar0 + kb + 32, 0, 1);
      __builtin_prefetch(wr[0] + kb + 32, 0, 1);
    }
    v16h a[2], b[4];
    a[0] = frag_from_f16(ar0 + kb, hh);
    a[1] = frag_from_f16(ar1 + kb, hh);
#pragma unroll
    for (int j = 0; j < 4; ++j) b[j] = frag_from_f16(wr[j] + kb, hh);
#pragma unroll
    for (int i = 0; i < 2; ++i)
#pragma unroll
      for (int j = 0; j < 4; ++j)
        acc[i][j] = wmma_f16(a[i], b[j], acc[i][j]);
  }

#pragma unroll
  for (int i = 0; i < 2; ++i)
#pragma unroll
    for (int j = 0; j < 4; ++j)
#pragma unroll
      for (int g = 0; g < 8; ++g) {
        const int m = Mb + 16 * i + g + 8 * hh;
        const int n = Nb + 16 * j + r;
        Out[(size_t)m * DMODEL + n] = acc[i][j][g];
      }
}

// ---------------------------------------------------------------------------
// Launch: f16 conversions -> QKV projections -> flash attention -> O proj.
// Workspace (f16 elems): Q 4M | K 4M | V^T 4M | Attn 4M | Xh 4M | 4x Wh 1M
//   = 24M halves = 48 MB. Whole working set lives in the 192 MB L2.
// ---------------------------------------------------------------------------
extern "C" void kernel_launch(void* const* d_in, const int* in_sizes, int n_in,
                              void* d_out, int out_size, void* d_ws, size_t ws_size,
                              hipStream_t stream) {
  const float* wq = (const float*)d_in[0];
  const float* wk = (const float*)d_in[1];
  const float* wv = (const float*)d_in[2];
  const float* wo = (const float*)d_in[3];
  const float* x  = (const float*)d_in[4];
  float* out = (float*)d_out;

  const size_t per  = (size_t)BATCHN * NUM_HEADS * SEQ * DHEAD;  // 4M elements
  const size_t wsz  = (size_t)DMODEL * DMODEL;                   // 1M elements
  _Float16* qw  = (_Float16*)d_ws;
  _Float16* kw  = qw + per;
  _Float16* vt  = kw + per;
  _Float16* at  = vt + per;   // BATCHN*SEQ*DMODEL == per
  _Float16* xh  = at + per;
  _Float16* wqh = xh + per;
  _Float16* wkh = wqh + wsz;
  _Float16* wvh = wkh + wsz;
  _Float16* woh = wvh + wsz;

  // one-shot f32 -> f16 conversions (removes all cvt work from GEMM loops)
  const int xn8 = (int)(per / 8), wn8 = (int)(wsz / 8);
  cvt_f16_kernel<<<(xn8 + 255) / 256, 256, 0, stream>>>(x, xh, xn8);
  cvt_f16_kernel<<<(wn8 + 255) / 256, 256, 0, stream>>>(wq, wqh, wn8);
  cvt_f16_kernel<<<(wn8 + 255) / 256, 256, 0, stream>>>(wk, wkh, wn8);
  cvt_f16_kernel<<<(wn8 + 255) / 256, 256, 0, stream>>>(wv, wvh, wn8);
  cvt_f16_kernel<<<(wn8 + 255) / 256, 256, 0, stream>>>(wo, woh, wn8);

  dim3 gproj(DMODEL / 128, (BATCHN * SEQ) / 128);  // (8, 32)
  proj_gemm_kernel<<<gproj, 256, 0, stream>>>(xh, wqh, qw, 0);
  proj_gemm_kernel<<<gproj, 256, 0, stream>>>(xh, wkh, kw, 0);
  proj_gemm_kernel<<<gproj, 256, 0, stream>>>(xh, wvh, vt, 1);

  dim3 gattn(SEQ / 64, BATCHN * NUM_HEADS);        // (32, 32)
  flash_attn_kernel<<<gattn, 128, 0, stream>>>(qw, kw, vt, at);

  out_gemm_kernel<<<gproj, 256, 0, stream>>>(at, woh, out);
}
